// SindyNetCompEnsemble_83966610636866
// MI455X (gfx1250) — compile-verified, matrix operand order checked by reading, not run on hardware
//
#include <hip/hip_runtime.h>

// ---------------------------------------------------------------------------
// Types / helpers
// ---------------------------------------------------------------------------
typedef __attribute__((ext_vector_type(16))) __bf16 v16bf;
typedef __attribute__((ext_vector_type(8)))  float  v8f;

typedef uint4  uint4_a  __attribute__((may_alias));
typedef uint2  uint2_a  __attribute__((may_alias));
typedef float4 float4_a __attribute__((may_alias));

union Frag {
  uint4 q[2];
  v16bf v;
};

__device__ __forceinline__ unsigned short f2bf(float f) {
  unsigned u = __float_as_uint(f);
  u += 0x7FFFu + ((u >> 16) & 1u);          // round-to-nearest-even
  return (unsigned short)(u >> 16);
}

__device__ __forceinline__ float sigmoidf(float x) {
  return 1.0f / (1.0f + __expf(-x));
}

// Accumulate K columns (multiple of 32) of a 16x16 tile:
//   aRow: row pointer for this lane's A row (M = lane&15), contiguous bf16
//   wRow: row pointer for this lane's B row (N = lane&15), contiguous bf16
// A 16-bit 16x32 fragment: lane (kh = lane>>4) holds K runs [k+kh*8 .. +8) and
// [k+16+kh*8 .. +8) -> two 16B loads each for A and B.
template<int K>
__device__ __forceinline__ v8f acc_chunk(v8f acc,
                                         const unsigned short* __restrict__ aRow,
                                         const unsigned short* __restrict__ wRow,
                                         int kh) {
#pragma unroll
  for (int k = 0; k < K; k += 32) {
    Frag a, b;
    const uint4_a* ap = (const uint4_a*)(aRow + k + kh * 8);
    const uint4_a* bp = (const uint4_a*)(wRow + k + kh * 8);
    a.q[0] = ap[0];  a.q[1] = ap[2];
    b.q[0] = bp[0];  b.q[1] = bp[2];
    acc = __builtin_amdgcn_wmma_f32_16x16x32_bf16(false, a.v, false, b.v,
                                                  (short)0, acc, false, false);
  }
  return acc;
}

// C/D layout: lane holds rows M = (lane>>4)*8 + v, col N = lane&15 (16x16 f32)
template<bool SIG, bool PAD16>
__device__ __forceinline__ void store_tile_lds(v8f acc, float bb,
                                               unsigned short* D, int ldD,
                                               int col, int kh) {
#pragma unroll
  for (int v = 0; v < 8; ++v) {
    float val = acc[v] + bb;
    if (SIG) val = sigmoidf(val);
    int r = kh * 8 + v;
    D[r * ldD + col] = f2bf(val);
    if (PAD16) D[r * ldD + col + 16] = 0;   // zero-pad so padded K is safe
  }
}

// Residual layer: out = sigmoid(W * [identity | act] + b), K split KZ + KA
template<int KZ, int KA, int NOUT, bool SIG, bool PAD16>
__device__ __forceinline__ void res_layer(
    const unsigned short* __restrict__ zRow,
    const unsigned short* __restrict__ actRow,
    const unsigned short* __restrict__ W, int ldW,
    const float* __restrict__ bias,
    unsigned short* D, int ldD, int m15, int kh) {
  for (int nt = 0; nt < NOUT / 16; ++nt) {
    const unsigned short* wRow = W + (long)(nt * 16 + m15) * ldW;
    v8f acc = {};
    acc = acc_chunk<KZ>(acc, zRow, wRow, kh);
    acc = acc_chunk<KA>(acc, actRow, wRow + KZ, kh);
    store_tile_lds<SIG, PAD16>(acc, bias[nt * 16 + m15], D, ldD,
                               nt * 16 + m15, kh);
  }
}

// Final residual layer: no sigmoid, bf16 store to global (row stride ldD)
template<int KZ, int KA, int NOUT>
__device__ __forceinline__ void res_layer_out(
    const unsigned short* __restrict__ zRow,
    const unsigned short* __restrict__ actRow,
    const unsigned short* __restrict__ W, int ldW,
    const float* __restrict__ bias,
    unsigned short* __restrict__ D, int ldD, int m15, int kh) {
  for (int nt = 0; nt < NOUT / 16; ++nt) {
    const unsigned short* wRow = W + (long)(nt * 16 + m15) * ldW;
    v8f acc = {};
    acc = acc_chunk<KZ>(acc, zRow, wRow, kh);
    acc = acc_chunk<KA>(acc, actRow, wRow + KZ, kh);
    float bb = bias[nt * 16 + m15];
    int col = nt * 16 + m15;
#pragma unroll
    for (int v = 0; v < 8; ++v)
      D[(long)(kh * 8 + v) * ldD + col] = f2bf(acc[v] + bb);
  }
}

#define WAVES 4
#define ROWS  (WAVES * 16)
#define BATCH 32768

// ---------------------------------------------------------------------------
// Weight f32 -> bf16 (with zero-padded K columns)
// ---------------------------------------------------------------------------
__global__ void conv_w(const float* __restrict__ src,
                       unsigned short* __restrict__ dst,
                       int cnt, int N, int K, int Kp) {
  long total = (long)cnt * N * Kp;
  for (long i = (long)blockIdx.x * blockDim.x + threadIdx.x; i < total;
       i += (long)gridDim.x * blockDim.x) {
    int  k  = (int)(i % Kp);
    long nm = i / Kp;
    float v = (k < K) ? src[nm * (long)K + k] : 0.0f;
    dst[i] = f2bf(v);
  }
}

// ---------------------------------------------------------------------------
// Encoder: per-encoder 256 -> 128 -> 64 -> 32 (sigmoid, sigmoid, none)
// grid (BATCH/ROWS, 16), block 128 (4 waves, 16 rows each, no barriers)
// ---------------------------------------------------------------------------
__global__ __launch_bounds__(128) void enc_kernel(
    const float* __restrict__ x,
    const unsigned short* __restrict__ w0, const float* __restrict__ b0,
    const unsigned short* __restrict__ w1, const float* __restrict__ b1,
    const unsigned short* __restrict__ w2, const float* __restrict__ b2,
    unsigned short* __restrict__ zs) {
  constexpr int L0 = 264, L1 = 136, L2 = 72;  // +8 skew vs 64-bank LDS
  __shared__ __align__(16) unsigned short sA0[WAVES][16 * L0];
  __shared__ __align__(16) unsigned short sH1[WAVES][16 * L1];
  __shared__ __align__(16) unsigned short sH2[WAVES][16 * L2];

  const int e    = blockIdx.y;
  const int wave = threadIdx.x >> 5;
  const int lane = threadIdx.x & 31;
  const int m15  = lane & 15;
  const int kh   = lane >> 4;
  const long row0 = (long)blockIdx.x * ROWS + wave * 16;

  unsigned short* A0 = sA0[wave];
  unsigned short* H1 = sH1[wave];
  unsigned short* H2 = sH2[wave];

  // Stage 16x256 f32 tile -> bf16 LDS (float4 loads, uint2 stores)
  {
    const float* src = x + row0 * 4096 + e * 256;
    for (int i = lane; i < 16 * 64; i += 32) {
      int r = i >> 6, c4 = i & 63;
      float4 vv = *(const float4_a*)(src + (long)r * 4096 + c4 * 4);
      uint2 p;
      p.x = (unsigned)f2bf(vv.x) | ((unsigned)f2bf(vv.y) << 16);
      p.y = (unsigned)f2bf(vv.z) | ((unsigned)f2bf(vv.w) << 16);
      *(uint2_a*)(A0 + r * L0 + c4 * 4) = p;
    }
  }

  const unsigned short* aRow0 = A0 + m15 * L0;
  const unsigned short* W0 = w0 + (long)e * 128 * 256;
  const float*          B0 = b0 + e * 128;
  for (int nt = 0; nt < 8; ++nt) {
    v8f acc = {};
    acc = acc_chunk<256>(acc, aRow0, W0 + (long)(nt * 16 + m15) * 256, kh);
    store_tile_lds<true, false>(acc, B0[nt * 16 + m15], H1, L1,
                                nt * 16 + m15, kh);
  }

  const unsigned short* aRow1 = H1 + m15 * L1;
  const unsigned short* W1 = w1 + (long)e * 64 * 128;
  const float*          B1 = b1 + e * 64;
  for (int nt = 0; nt < 4; ++nt) {
    v8f acc = {};
    acc = acc_chunk<128>(acc, aRow1, W1 + (long)(nt * 16 + m15) * 128, kh);
    store_tile_lds<true, false>(acc, B1[nt * 16 + m15], H2, L2,
                                nt * 16 + m15, kh);
  }

  const unsigned short* aRow2 = H2 + m15 * L2;
  const unsigned short* W2 = w2 + (long)e * 32 * 64;
  const float*          B2 = b2 + e * 32;
  unsigned short* dst = zs + row0 * 512 + e * 32;
  for (int nt = 0; nt < 2; ++nt) {
    v8f acc = {};
    acc = acc_chunk<64>(acc, aRow2, W2 + (long)(nt * 16 + m15) * 64, kh);
    float bb = B2[nt * 16 + m15];
    int col = nt * 16 + m15;
#pragma unroll
    for (int v = 0; v < 8; ++v)
      dst[(long)(kh * 8 + v) * 512 + col] = f2bf(acc[v] + bb);
  }
}

// ---------------------------------------------------------------------------
// Compressor: identity width 512, residual layers; z read from global (L2-hot)
// ---------------------------------------------------------------------------
__global__ __launch_bounds__(128) void comp_kernel(
    const unsigned short* __restrict__ zs,
    const unsigned short* __restrict__ w0, const float* __restrict__ b0,
    const unsigned short* __restrict__ w1, const float* __restrict__ b1,
    const unsigned short* __restrict__ w2, const float* __restrict__ b2,
    const unsigned short* __restrict__ w3, const float* __restrict__ b3,
    const unsigned short* __restrict__ w4, const float* __restrict__ b4,
    const unsigned short* __restrict__ w5, const float* __restrict__ b5,
    unsigned short* __restrict__ zo) {
  constexpr int LA = 264, LB = 136;
  __shared__ __align__(16) unsigned short sA[WAVES][16 * LA];
  __shared__ __align__(16) unsigned short sB[WAVES][16 * LB];

  const int wave = threadIdx.x >> 5;
  const int lane = threadIdx.x & 31;
  const int m15  = lane & 15;
  const int kh   = lane >> 4;
  const long row0 = (long)blockIdx.x * ROWS + wave * 16;

  unsigned short* bufA = sA[wave];
  unsigned short* bufB = sB[wave];
  const unsigned short* zRow = zs + (row0 + m15) * 512;  // global identity row
  const unsigned short* aA = bufA + m15 * LA;
  const unsigned short* aB = bufB + m15 * LB;

  res_layer<512, 512, 256, true, false>(zRow, zRow, w0, 1024, b0, bufA, LA, m15, kh);
  res_layer<512, 256, 128, true, false>(zRow, aA,   w1,  768, b1, bufB, LB, m15, kh);
  res_layer<512, 128,  64, true, false>(zRow, aB,   w2,  640, b2, bufA, LA, m15, kh);
  res_layer<512,  64,  32, true, false>(zRow, aA,   w3,  576, b3, bufB, LB, m15, kh);
  res_layer<512,  32,  16, true, true >(zRow, aB,   w4,  544, b4, bufA, LA, m15, kh);
  res_layer_out<512, 32, 32>(zRow, aA, w5, 544, b5, zo + row0 * 32, 32, m15, kh);
}

// ---------------------------------------------------------------------------
// Decompressor: identity width 32 (staged in LDS)
// ---------------------------------------------------------------------------
__global__ __launch_bounds__(128) void decomp_kernel(
    const unsigned short* __restrict__ zi,
    const unsigned short* __restrict__ w0, const float* __restrict__ b0,
    const unsigned short* __restrict__ w1, const float* __restrict__ b1,
    const unsigned short* __restrict__ w2, const float* __restrict__ b2,
    const unsigned short* __restrict__ w3, const float* __restrict__ b3,
    const unsigned short* __restrict__ w4, const float* __restrict__ b4,
    const unsigned short* __restrict__ w5, const float* __restrict__ b5,
    unsigned short* __restrict__ zsh) {
  constexpr int LZ = 40, LA = 264, LB = 136;
  __shared__ __align__(16) unsigned short sZ[WAVES][16 * LZ];
  __shared__ __align__(16) unsigned short sA[WAVES][16 * LA];
  __shared__ __align__(16) unsigned short sB[WAVES][16 * LB];

  const int wave = threadIdx.x >> 5;
  const int lane = threadIdx.x & 31;
  const int m15  = lane & 15;
  const int kh   = lane >> 4;
  const long row0 = (long)blockIdx.x * ROWS + wave * 16;

  unsigned short* Z    = sZ[wave];
  unsigned short* bufA = sA[wave];
  unsigned short* bufB = sB[wave];

  {
    const unsigned short* src = zi + row0 * 32;
    for (int i = lane; i < 64; i += 32) {
      int r = i >> 2, c = i & 3;
      *(uint4_a*)(Z + r * LZ + c * 8) = *(const uint4_a*)(src + (long)r * 32 + c * 8);
    }
  }

  const unsigned short* zRow = Z + m15 * LZ;
  const unsigned short* aA = bufA + m15 * LA;
  const unsigned short* aB = bufB + m15 * LB;

  res_layer<32,  32, 256, true, false>(zRow, zRow, w0,  64, b0, bufA, LA, m15, kh);
  res_layer<32, 256, 128, true, false>(zRow, aA,   w1, 288, b1, bufB, LB, m15, kh);
  res_layer<32, 128,  64, true, false>(zRow, aB,   w2, 160, b2, bufA, LA, m15, kh);
  res_layer<32,  64,  32, true, false>(zRow, aA,   w3,  96, b3, bufB, LB, m15, kh);
  res_layer<32,  32,  16, true, true >(zRow, aB,   w4,  64, b4, bufA, LA, m15, kh);
  res_layer_out<32, 32, 512>(zRow, aA, w5, 64, b5, zsh + row0 * 512, 512, m15, kh);
}

// ---------------------------------------------------------------------------
// Decoder: per-encoder 32 -> 64 -> 128 -> 256, final f32 to d_out
// ---------------------------------------------------------------------------
__global__ __launch_bounds__(128) void dec_kernel(
    const unsigned short* __restrict__ zsh,
    const unsigned short* __restrict__ w0, const float* __restrict__ b0,
    const unsigned short* __restrict__ w1, const float* __restrict__ b1,
    const unsigned short* __restrict__ w2, const float* __restrict__ b2,
    float* __restrict__ out) {
  constexpr int LZ = 40, L1 = 72, L2 = 136;
  __shared__ __align__(16) unsigned short sZ[WAVES][16 * LZ];
  __shared__ __align__(16) unsigned short sH1[WAVES][16 * L1];
  __shared__ __align__(16) unsigned short sH2[WAVES][16 * L2];

  const int e    = blockIdx.y;
  const int wave = threadIdx.x >> 5;
  const int lane = threadIdx.x & 31;
  const int m15  = lane & 15;
  const int kh   = lane >> 4;
  const long row0 = (long)blockIdx.x * ROWS + wave * 16;

  unsigned short* Z  = sZ[wave];
  unsigned short* H1 = sH1[wave];
  unsigned short* H2 = sH2[wave];

  {
    const unsigned short* src = zsh + row0 * 512 + e * 32;
    for (int i = lane; i < 64; i += 32) {
      int r = i >> 2, c = i & 3;
      *(uint4_a*)(Z + r * LZ + c * 8) = *(const uint4_a*)(src + (long)r * 512 + c * 8);
    }
  }

  const unsigned short* aRow0 = Z + m15 * LZ;
  const unsigned short* W0 = w0 + (long)e * 64 * 32;
  const float*          B0 = b0 + e * 64;
  for (int nt = 0; nt < 4; ++nt) {
    v8f acc = {};
    acc = acc_chunk<32>(acc, aRow0, W0 + (long)(nt * 16 + m15) * 32, kh);
    store_tile_lds<true, false>(acc, B0[nt * 16 + m15], H1, L1, nt * 16 + m15, kh);
  }

  const unsigned short* aRow1 = H1 + m15 * L1;
  const unsigned short* W1 = w1 + (long)e * 128 * 64;
  const float*          B1 = b1 + e * 128;
  for (int nt = 0; nt < 8; ++nt) {
    v8f acc = {};
    acc = acc_chunk<64>(acc, aRow1, W1 + (long)(nt * 16 + m15) * 64, kh);
    store_tile_lds<true, false>(acc, B1[nt * 16 + m15], H2, L2, nt * 16 + m15, kh);
  }

  const unsigned short* aRow2 = H2 + m15 * L2;
  const unsigned short* W2 = w2 + (long)e * 256 * 128;
  const float*          B2 = b2 + e * 256;
  float* dst = out + row0 * 4096 + e * 256;
  for (int nt = 0; nt < 16; ++nt) {
    v8f acc = {};
    acc = acc_chunk<128>(acc, aRow2, W2 + (long)(nt * 16 + m15) * 128, kh);
    float bb = B2[nt * 16 + m15];
    int col = nt * 16 + m15;
#pragma unroll
    for (int v = 0; v < 8; ++v)
      dst[(long)(kh * 8 + v) * 4096 + col] = acc[v] + bb;
  }
}

// ---------------------------------------------------------------------------
// Host launch
// ---------------------------------------------------------------------------
extern "C" void kernel_launch(void* const* d_in, const int* in_sizes, int n_in,
                              void* d_out, int out_size, void* d_ws, size_t ws_size,
                              hipStream_t stream) {
  (void)in_sizes; (void)n_in; (void)out_size; (void)ws_size;

  const float* x = (const float*)d_in[0];
  const float* enc_w[3] = {(const float*)d_in[1], (const float*)d_in[3], (const float*)d_in[5]};
  const float* enc_b[3] = {(const float*)d_in[2], (const float*)d_in[4], (const float*)d_in[6]};
  const float* dec_w[3] = {(const float*)d_in[7], (const float*)d_in[9], (const float*)d_in[11]};
  const float* dec_b[3] = {(const float*)d_in[8], (const float*)d_in[10], (const float*)d_in[12]};
  const float* cw[6]; const float* cb[6];
  const float* pw[6]; const float* pb[6];
  for (int i = 0; i < 6; ++i) {
    cw[i] = (const float*)d_in[13 + 2 * i];  cb[i] = (const float*)d_in[14 + 2 * i];
    pw[i] = (const float*)d_in[25 + 2 * i];  pb[i] = (const float*)d_in[26 + 2 * i];
  }

  char* ws = (char*)d_ws;
  size_t off = 0;
  auto take = [&](size_t elems) -> size_t {
    size_t o = off;
    off += ((elems * 2 + 255) & ~(size_t)255);
    return o;
  };
  const size_t B = BATCH;
  size_t o_zs  = take(B * 512);
  size_t o_z   = take(B * 32);
  size_t o_zsh = take(B * 512);

  size_t o_ew[3] = {take(16ull*128*256), take(16ull*64*128), take(16ull*32*64)};
  size_t o_dw[3] = {take(16ull*64*32),   take(16ull*128*64), take(16ull*256*128)};
  const int cN[6] = {256,128,64,32,16,32},  cK[6] = {1024,768,640,576,544,528},
            cKp[6]= {1024,768,640,576,544,544};
  const int pN[6] = {256,128,64,32,16,512}, pK[6] = {64,288,160,96,64,48},
            pKp[6]= {64,288,160,96,64,64};
  size_t o_cw[6], o_pw[6];
  for (int i = 0; i < 6; ++i) o_cw[i] = take((size_t)cN[i] * cKp[i]);
  for (int i = 0; i < 6; ++i) o_pw[i] = take((size_t)pN[i] * pKp[i]);

  auto W = [&](size_t o) { return (unsigned short*)(ws + o); };
  auto conv = [&](const float* src, size_t dst, int cnt, int N, int K, int Kp) {
    long total = (long)cnt * N * Kp;
    int blocks = (int)((total + 255) / 256);
    if (blocks > 2048) blocks = 2048;
    conv_w<<<blocks, 256, 0, stream>>>(src, W(dst), cnt, N, K, Kp);
  };

  conv(enc_w[0], o_ew[0], 16, 128, 256, 256);
  conv(enc_w[1], o_ew[1], 16,  64, 128, 128);
  conv(enc_w[2], o_ew[2], 16,  32,  64,  64);
  conv(dec_w[0], o_dw[0], 16,  64,  32,  32);
  conv(dec_w[1], o_dw[1], 16, 128,  64,  64);
  conv(dec_w[2], o_dw[2], 16, 256, 128, 128);
  for (int i = 0; i < 6; ++i) conv(cw[i], o_cw[i], 1, cN[i], cK[i], cKp[i]);
  for (int i = 0; i < 6; ++i) conv(pw[i], o_pw[i], 1, pN[i], pK[i], pKp[i]);

  dim3 egrid(BATCH / ROWS, 16);
  enc_kernel<<<egrid, 128, 0, stream>>>(
      x, W(o_ew[0]), enc_b[0], W(o_ew[1]), enc_b[1], W(o_ew[2]), enc_b[2],
      W(o_zs));

  comp_kernel<<<BATCH / ROWS, 128, 0, stream>>>(
      W(o_zs),
      W(o_cw[0]), cb[0], W(o_cw[1]), cb[1], W(o_cw[2]), cb[2],
      W(o_cw[3]), cb[3], W(o_cw[4]), cb[4], W(o_cw[5]), cb[5],
      W(o_z));

  decomp_kernel<<<BATCH / ROWS, 128, 0, stream>>>(
      W(o_z),
      W(o_pw[0]), pb[0], W(o_pw[1]), pb[1], W(o_pw[2]), pb[2],
      W(o_pw[3]), pb[3], W(o_pw[4]), pb[4], W(o_pw[5]), pb[5],
      W(o_zsh));

  dec_kernel<<<egrid, 128, 0, stream>>>(
      W(o_zsh), W(o_dw[0]), dec_b[0], W(o_dw[1]), dec_b[1], W(o_dw[2]), dec_b[2],
      (float*)d_out);
}
